// TERMNeighborAttention_64673617543289
// MI455X (gfx1250) — matrix operand order repaired and checked by reading, not compiled, not run on hardware
//
#include <hip/hip_runtime.h>
#include <hip/hip_bf16.h>

// -------- problem constants (from reference setup_inputs) ----------
#define BB 8
#define TT 64
#define NN 24
#define KK 30
#define HH 128
#define IN 256
#define NGROUPS (BB * TT * NN)   // 12288
#define NHEADS 4
#define DHEAD 32                 // H / NHEADS
#define NEG_INF_F (-3.4028234663852886e38f)
#define INV_SQRT_D 0.17677669529663687f

typedef __bf16 bf16_t;
typedef bf16_t v16bf __attribute__((ext_vector_type(16)));
typedef float  v8f   __attribute__((ext_vector_type(8)));
typedef unsigned int v4u __attribute__((ext_vector_type(4)));
typedef int v4i __attribute__((ext_vector_type(4)));
typedef int v8i __attribute__((ext_vector_type(8)));

union Frag16 { v16bf v; unsigned short u[16]; };

// Dynamic LDS layout (~115 KB, fits 320 KB/WGP).
// sW doubles as the fp32 TDM staging buffer for the h_EV tile (30 KB needed,
// 64 KB available) before the bf16 weights are loaded into it.
struct Smem {
    unsigned short sW[HH * IN];   // weights bf16 [n][k] row-major (64 KB) / fp32 staging
    unsigned short sA[32 * IN];   // h_EV tile bf16, rows 30..31 zero-padded (16 KB)
    float sKp[32 * HH];           // K projection f32 (16 KB)
    float sV [32 * HH];           // V projection f32 (16 KB)
    float shV[HH];
    float sQ [HH];
    float shU[HH];
    float sLog[NHEADS * 32];
    float sAtt[NHEADS * 32];
};

__device__ __forceinline__ unsigned short f2bf_rne(float f) {
    unsigned int u = __float_as_uint(f);
    unsigned int r = u + 0x7FFFu + ((u >> 16) & 1u);
    return (unsigned short)(r >> 16);
}
__device__ __forceinline__ unsigned int pack2bf(float a, float b) {
    return (unsigned int)f2bf_rne(a) | ((unsigned int)f2bf_rne(b) << 16);
}

__global__ void term_neighbor_attn_kernel(const float* __restrict__ h_V,
                                          const float* __restrict__ h_EV,
                                          const int*   __restrict__ mask,
                                          const float* __restrict__ W_Q,
                                          const float* __restrict__ W_K,
                                          const float* __restrict__ W_V,
                                          const float* __restrict__ W_O,
                                          float* __restrict__ out) {
    extern __shared__ char smem_raw[];
    Smem* S = reinterpret_cast<Smem*>(smem_raw);

    const int g    = blockIdx.x;                 // (b,t,n) flat group
    const int tid  = threadIdx.x;                // 0..255 (8 wave32)
    const int lane = tid & 31;
    const int wv   = tid >> 5;
    const int lrow = lane & 15;                  // row/col within 16-wide tile
    const int lhi  = (lane >> 4) & 1;            // lane half select

    const float* hv  = h_V  + (size_t)g * HH;
    const float* hev = h_EV + (size_t)g * (KK * IN);
    const int*   msk = mask + (size_t)g * KK;

    __builtin_prefetch(W_K, 0, 0);   // global_prefetch_b8 — weights are L2-resident
    __builtin_prefetch(W_V, 0, 0);

    // ---- Phase 0a: TDM async load of the fp32 h_EV tile (30 rows x 256 f32)
    //      into the LDS staging area (offset 0, aliases sW). Wave 0 issues it.
    if (tid == 0) {
        unsigned long long ga = (unsigned long long)(uintptr_t)hev;
        v4u g0;
        // [1:0] count=1, [2] is_restore=0, [31:30] gather off
        g0[0] = 1u;
        // [63:32] lds_addr = 0 (staging at LDS base)
        g0[1] = 0u;
        // [120:64] global_addr, [127:126] type=2 ("image")
        g0[2] = (unsigned int)ga;
        g0[3] = (unsigned int)(ga >> 32) | (2u << 30);

        v8i g1;
        g1[0] = 0x00020000;                // workgroup_mask=0, data_size=2 (4 bytes)
        g1[1] = (int)(256u << 16);         // tensor_dim0[15:0]=256 at bits[63:48]
        g1[2] = (int)((unsigned)KK << 16); // tensor_dim1[15:0]=30 at bits[95:80]
        g1[3] = (int)(256u << 16);         // tile_dim0=256 at bits[127:112]
        g1[4] = KK;                        // tile_dim1=30 at bits[143:128]
        g1[5] = 256;                       // tensor_dim0_stride[31:0]=256
        g1[6] = 0;
        g1[7] = 0;

        v4i z4 = {};
#if __has_include(<hip/amd_detail/amd_gfx1250_TDM.h>)
        v8i z8 = {};
        __builtin_amdgcn_tensor_load_to_lds(g0, g1, z4, z4, z8, 0);
#else
        __builtin_amdgcn_tensor_load_to_lds(g0, g1, z4, z4, 0);
#endif
        __builtin_amdgcn_s_wait_tensorcnt(0);
    }
    if (tid < HH) S->shV[tid] = hv[tid];
    __syncthreads();   // staging tile + shV visible to all

    // ---- Phase 0b: convert staged fp32 tile -> bf16 sA (pad rows 30..31) ----
    {
        const float4* stg = reinterpret_cast<const float4*>(S->sW);
        uint2* dst = reinterpret_cast<uint2*>(S->sA);
        for (int i = tid; i < 32 * IN / 4; i += 256) {
            uint2 w;
            if (i < KK * IN / 4) {
                float4 v = stg[i];
                w.x = pack2bf(v.x, v.y);
                w.y = pack2bf(v.z, v.w);
            } else {
                w.x = 0u; w.y = 0u;
            }
            dst[i] = w;
        }
    }

    // ---- Phase 1: Q = W_Q @ h_v_row (small matvec, fp32 VALU, b128 loads) ----
    if (tid < HH) {
        const float4* wr = reinterpret_cast<const float4*>(W_Q + (size_t)tid * HH);
        const float4* xv = reinterpret_cast<const float4*>(S->shV);
        float acc = 0.f;
        #pragma unroll 8
        for (int k = 0; k < HH / 4; ++k) {
            float4 w = wr[k];
            float4 x = xv[k];
            acc += w.x * x.x + w.y * x.y + w.z * x.z + w.w * x.w;
        }
        S->sQ[tid] = acc;
    }

    // ---- Phase 2: Kp / V projections via v_wmma_f32_16x16x32_bf16 ----
    // D(32x128) = A(32x256 bf16) x B(256x128), B[k][n] = W[n][k].
    // Each wave owns one 16-wide N tile (nt = wave id) and runs TWO
    // independent accumulator chains (mt=0 / mt=1) sharing the B fragment.
    // Fragments are double-buffered: loads for k-step ks+1 issue before the
    // WMMAs for ks, so the waitcnt pass can leave the prefetch outstanding
    // (partial s_wait_dscnt) and LDS latency overlaps the matrix math.
    for (int proj = 0; proj < 2; ++proj) {
        __syncthreads();   // prior consumers of sW (staging / previous proj) done
        {
            const float4* wsrc = reinterpret_cast<const float4*>(proj == 0 ? W_K : W_V);
            uint2* wdst = reinterpret_cast<uint2*>(S->sW);
            for (int i = tid; i < HH * IN / 4; i += 256) {
                float4 v = wsrc[i];
                uint2 w;
                w.x = pack2bf(v.x, v.y);
                w.y = pack2bf(v.z, v.w);
                wdst[i] = w;
            }
        }
        __syncthreads();

        float* Dst = (proj == 0) ? S->sKp : S->sV;
        const int nt = wv;   // 8 waves -> nt 0..7
        const unsigned short* aBase0 = S->sA + lrow * IN;          // mt = 0
        const unsigned short* aBase1 = S->sA + (16 + lrow) * IN;   // mt = 1
        const unsigned short* bBase  = S->sW + (nt * 16 + lrow) * IN;
        const int aoff = lhi * 8;    // A: lanes 0-15 get K 0..7/16..23, 16-31 get 8..15/24..31
        const int boff = lhi * 8;    // B: lanes 0-15 hold K 0..7/16..23, 16-31 hold 8..15/24..31

        Frag16 a0[2], a1[2], b[2];
        #define LOAD_FRAGS(ks, buf)                                          \
            do {                                                             \
                const int ka_ = (ks) * 32;                                   \
                _Pragma("unroll")                                            \
                for (int i = 0; i < 8; ++i) {                                \
                    a0[buf].u[i]     = aBase0[ka_ + aoff + i];               \
                    a0[buf].u[8 + i] = aBase0[ka_ + aoff + 16 + i];          \
                    a1[buf].u[i]     = aBase1[ka_ + aoff + i];               \
                    a1[buf].u[8 + i] = aBase1[ka_ + aoff + 16 + i];          \
                }                                                            \
                _Pragma("unroll")                                            \
                for (int i = 0; i < 8; ++i) { b[buf].u[i] = bBase[ka_ + boff + i]; b[buf].u[8 + i] = bBase[ka_ + boff + 16 + i]; } \
            } while (0)

        LOAD_FRAGS(0, 0);
        v8f acc0 = {};
        v8f acc1 = {};
        #pragma unroll
        for (int ks = 0; ks < 8; ++ks) {
            const int cur = ks & 1;
            if (ks < 7) LOAD_FRAGS(ks + 1, cur ^ 1);   // prefetch next k-step
            acc0 = __builtin_amdgcn_wmma_f32_16x16x32_bf16(
                     false, a0[cur].v, false, b[cur].v, (short)0, acc0, false, false);
            acc1 = __builtin_amdgcn_wmma_f32_16x16x32_bf16(
                     false, a1[cur].v, false, b[cur].v, (short)0, acc1, false, false);
        }
        #undef LOAD_FRAGS

        // D layout: lane half selects M+8, VGPR j is M row offset
        const int dcol = nt * 16 + lrow;
        #pragma unroll
        for (int j = 0; j < 8; ++j) {
            Dst[(lhi * 8 + j) * HH + dcol]      = acc0[j];   // mt=0 rows 0..15
            Dst[(16 + lhi * 8 + j) * HH + dcol] = acc1[j];   // mt=1 rows 16..31
        }
    }
    __syncthreads();

    // ---- Phase 3: masked logits  logits[h][k] = (Q_h . Kp[k]_h) / sqrt(d) ----
    if (tid < NHEADS * KK) {
        const int h = tid / KK;
        const int k = tid - h * KK;
        const float* qh = S->sQ + h * DHEAD;
        const float* kp = S->sKp + k * HH + h * DHEAD;
        float acc = 0.f;
        #pragma unroll 8
        for (int j = 0; j < DHEAD; ++j) acc += qh[j] * kp[j];
        S->sLog[h * 32 + k] = (msk[k] > 0) ? acc * INV_SQRT_D : NEG_INF_F;
    }
    __syncthreads();

    // ---- Phase 4: softmax over K per head, then zero masked lanes ----
    if (tid < NHEADS) {
        const int h = tid;
        float m = NEG_INF_F;
        for (int k = 0; k < KK; ++k) m = fmaxf(m, S->sLog[h * 32 + k]);
        float s = 0.f;
        float e[KK];
        for (int k = 0; k < KK; ++k) { e[k] = __expf(S->sLog[h * 32 + k] - m); s += e[k]; }
        const float inv = 1.f / s;
        for (int k = 0; k < KK; ++k)
            S->sAtt[h * 32 + k] = (msk[k] > 0) ? e[k] * inv : 0.f;
    }
    __syncthreads();

    // ---- Phase 5: h_upd[h*d + j] = sum_k att[h][k] * V[k][h*d + j] ----
    if (tid < HH) {
        const int h = tid >> 5;   // DHEAD == 32
        float acc = 0.f;
        #pragma unroll 6
        for (int k = 0; k < KK; ++k) acc += S->sAtt[h * 32 + k] * S->sV[k * HH + tid];
        S->shU[tid] = acc;
    }
    __syncthreads();

    // ---- Phase 6: out_row = W_O @ h_upd (b128 loads) ----
    if (tid < HH) {
        const float4* wr = reinterpret_cast<const float4*>(W_O + (size_t)tid * HH);
        const float4* xv = reinterpret_cast<const float4*>(S->shU);
        float acc = 0.f;
        #pragma unroll 8
        for (int k = 0; k < HH / 4; ++k) {
            float4 w = wr[k];
            float4 x = xv[k];
            acc += w.x * x.x + w.y * x.y + w.z * x.z + w.w * x.w;
        }
        out[(size_t)g * HH + tid] = acc;
    }
}

extern "C" void kernel_launch(void* const* d_in, const int* in_sizes, int n_in,
                              void* d_out, int out_size, void* d_ws, size_t ws_size,
                              hipStream_t stream) {
    const float* h_V  = (const float*)d_in[0];
    const float* h_EV = (const float*)d_in[1];
    const int*   mask = (const int*)  d_in[2];
    const float* W_Q  = (const float*)d_in[3];
    const float* W_K  = (const float*)d_in[4];
    const float* W_V  = (const float*)d_in[5];
    const float* W_O  = (const float*)d_in[6];
    float* out = (float*)d_out;

    dim3 grid(NGROUPS);
    dim3 block(256);
    size_t smem = sizeof(Smem);
    hipLaunchKernelGGL(term_neighbor_attn_kernel, grid, block, smem, stream,
                       h_V, h_EV, mask, W_Q, W_K, W_V, W_O, out);
}
